// WindowAttention_67156108640661
// MI455X (gfx1250) — matrix operand order repaired and checked by reading, not compiled, not run on hardware
//
#include <hip/hip_runtime.h>
#include <hip/hip_bf16.h>

typedef __attribute__((ext_vector_type(16))) _Float16 v16h;
typedef __attribute__((ext_vector_type(8)))  _Float16 v8h;
typedef __attribute__((ext_vector_type(8)))  float    v8f;

#define NWIN   1024          // 16 batches * 8 * 8 windows
#define NTOK   49            // 7x7 window
#define MROWS  (NWIN * NTOK) // 50176, multiple of 128
#define CDIM   512
#define QKVDIM 1536
#define NHEAD  16
#define HDIM   32
#define QSCALE 0.17677669529663687f  // 32^-0.5

// ---------------------------------------------------------------------------
// WMMA fragment loaders (A/B use this file's contiguous k order, not the ISA table order)
// A (16x32 f16):  lane<16: row=lane,    K = 0..15 contiguous
//                 lane>=16: row=lane-16, K = 16..31 contiguous
// B (32x16 f16, loaded from N-major storage Bt[n][k]):
//                 lane<16: col=lane,    K = 0..15 contiguous
//                 lane>=16: col=lane-16, K = 16..31 contiguous
// C/D (16x16 f32): lane<16: N=lane, M=vgpr;  lane>=16: N=lane-16, M=vgpr+8
// ---------------------------------------------------------------------------
__device__ __forceinline__ v16h load_frag_a(const _Float16* __restrict__ base,
                                            int ld, int lane) {
    int row = lane & 15;
    int kb  = (lane >> 4) << 4;           // 0 or 16
    const _Float16* p = base + row * ld + kb;
    union { v16h v; v8h h[2]; } u;
    u.h[0] = *(const v8h*)(p);
    u.h[1] = *(const v8h*)(p + 8);
    return u.v;
}

__device__ __forceinline__ v16h load_frag_b(const _Float16* __restrict__ baseT,
                                            int ld, int lane) {
    int n  = lane & 15;
    int kb = (lane >> 4) << 4;            // 0 or 16
    return *(const v16h*)(baseT + n * ld + kb);
}

__device__ __forceinline__ v8f wmma_f16(v16h a, v16h b, v8f c) {
    return __builtin_amdgcn_wmma_f32_16x16x32_f16(false, a, false, b,
                                                  (short)0, c, false, false);
}

// ---------------------------------------------------------------------------
// Kernel 0: transpose weights to N-major f16; fold q-scale into first 512 cols
// ---------------------------------------------------------------------------
__global__ __launch_bounds__(256) void prep_weights(
        const float* __restrict__ wqkv,   // (512, 1536) row-major (c, d)
        const float* __restrict__ wproj,  // (512, 512)  row-major (c, d)
        _Float16* __restrict__ wqkvT,     // (1536, 512) row-major (d, c)
        _Float16* __restrict__ wprojT) {  // (512, 512)  row-major (d, c)
    int idx = blockIdx.x * 256 + threadIdx.x;
    if (idx < QKVDIM * CDIM) {
        int d = idx >> 9;                 // output row
        int c = idx & 511;
        float v = wqkv[(size_t)c * QKVDIM + d];
        if (d < CDIM) v *= QSCALE;        // fold q scale into Wq
        wqkvT[idx] = (_Float16)v;
    }
    if (idx < CDIM * CDIM) {
        int d = idx >> 9;
        int c = idx & 511;
        wprojT[idx] = (_Float16)wproj[(size_t)c * CDIM + d];
    }
}

// ---------------------------------------------------------------------------
// Kernel 1: window-partition gather + fp32 -> f16 (vectorized: 2x float4 in,
// one 16B v8h out per thread)
// ---------------------------------------------------------------------------
__global__ __launch_bounds__(256) void gather_windows(
        const float* __restrict__ hs,     // (16, 3136, 512)
        _Float16* __restrict__ Xw) {      // (50176, 512)
    size_t idx = (size_t)blockIdx.x * 256 + threadIdx.x; // MROWS*64 total
    int r  = (int)(idx >> 6);
    int cb = ((int)idx & 63) << 3;        // 8 channels per thread
    int win = r / NTOK;
    int t   = r - win * NTOK;
    int b   = win >> 6;
    int wrem = win & 63;
    int wh = wrem >> 3, ww = wrem & 7;
    int i = t / 7, j = t - i * 7;
    int srow = (wh * 7 + i) * 56 + ww * 7 + j;
    const float4* src =
        (const float4*)(hs + ((size_t)b * 3136 + srow) * CDIM + cb);
    float4 a0 = src[0];
    float4 a1 = src[1];
    v8h o;
    o[0] = (_Float16)a0.x; o[1] = (_Float16)a0.y;
    o[2] = (_Float16)a0.z; o[3] = (_Float16)a0.w;
    o[4] = (_Float16)a1.x; o[5] = (_Float16)a1.y;
    o[6] = (_Float16)a1.z; o[7] = (_Float16)a1.w;
    *(v8h*)(Xw + (size_t)r * CDIM + cb) = o;
}

// ---------------------------------------------------------------------------
// Kernel 2: QKV GEMM  (50176 x 512) @ (512 x 1536) -> f16 (50176 x 1536)
// 128 threads = 4 waves, each wave computes a 64x64 tile (4x4 WMMA frags)
// block tile 128x128 ; grid (392, 12). Epilogue repacks the wave-private
// tile through LDS for coalesced b128 stores (barrier-free: same-wave LDS
// ops are in-order).
// ---------------------------------------------------------------------------
__global__ __launch_bounds__(128) void gemm_qkv(
        const _Float16* __restrict__ X,   // (M, 512)
        const _Float16* __restrict__ WT,  // (1536, 512) N-major
        _Float16* __restrict__ Y) {       // (M, 1536)
    __shared__ __align__(16) _Float16 stage[4][64 * 64];  // 8KB per wave
    int lane = threadIdx.x & 31;
    int wave = threadIdx.x >> 5;
    int m0 = blockIdx.x * 128 + (wave >> 1) * 64;
    int n0 = blockIdx.y * 128 + (wave & 1) * 64;

    v8f acc[4][4] = {};
#pragma unroll 4
    for (int k = 0; k < CDIM; k += 32) {
        v16h a[4], b[4];
#pragma unroll
        for (int i = 0; i < 4; i++)
            a[i] = load_frag_a(X + (size_t)(m0 + i * 16) * CDIM + k, CDIM, lane);
#pragma unroll
        for (int j = 0; j < 4; j++)
            b[j] = load_frag_b(WT + (size_t)(n0 + j * 16) * CDIM + k, CDIM, lane);
#pragma unroll
        for (int i = 0; i < 4; i++)
#pragma unroll
            for (int j = 0; j < 4; j++)
                acc[i][j] = wmma_f16(a[i], b[j], acc[i][j]);
    }

    // stage wave tile in LDS (D-fragment layout -> row-major 64x64)
    _Float16* st = stage[wave];
    int rowoff = (lane >> 4) << 3;
    int col    = lane & 15;
#pragma unroll
    for (int i = 0; i < 4; i++)
#pragma unroll
        for (int j = 0; j < 4; j++)
#pragma unroll
            for (int v = 0; v < 8; v++)
                st[(i * 16 + rowoff + v) * 64 + j * 16 + col] =
                    (_Float16)acc[i][j][v];

    // coalesced copy-out: 512 x 16B chunks, 32 lanes x 16 iters
#pragma unroll
    for (int it = 0; it < 16; it++) {
        int li = it * 32 + lane;          // 0..511
        int r  = li >> 3;                 // row 0..63
        int ch = li & 7;                  // 8-half chunk
        v8h d = *(v8h*)(st + r * 64 + ch * 8);
        *(v8h*)(Y + (size_t)(m0 + r) * QKVDIM + n0 + ch * 8) = d;
    }
}

// ---------------------------------------------------------------------------
// Kernel 3: windowed attention, one block per (window, head), 128 thr = 4 waves
// ---------------------------------------------------------------------------
#define SROW 65   // f32 score row stride (bank-conflict padding)
#define PROW 72   // f16 prob row stride (keeps 16B alignment for frag loads)

__global__ __launch_bounds__(128) void attn_win(
        const _Float16* __restrict__ QKV, // (M, 1536)
        _Float16* __restrict__ O) {       // (M, 512)
    __shared__ __align__(64) _Float16 Qs[64 * 32];
    __shared__ __align__(64) _Float16 Ks[64 * 32];
    __shared__ __align__(64) _Float16 Vt[32 * 64];   // transposed: Vt[e][tok]
    __shared__ __align__(64) float    S[64 * SROW];
    __shared__ __align__(64) _Float16 P[64 * PROW];

    int win = blockIdx.x;
    int h   = blockIdx.y;
    size_t rbase = (size_t)win * NTOK;
    int tid  = threadIdx.x;
    int lane = tid & 31;
    int wave = tid >> 5;

    // ---- stage zero-padded Q/K/V tiles (64 tokens x 32 ch) into LDS ----
    {
        int row = tid >> 1;
        int cb  = (tid & 1) * 16;
        if (row < NTOK) {
            const _Float16* src = QKV + (rbase + row) * QKVDIM + h * HDIM + cb;
            *(v8h*)(Qs + row * 32 + cb)     = *(const v8h*)(src);
            *(v8h*)(Qs + row * 32 + cb + 8) = *(const v8h*)(src + 8);
            *(v8h*)(Ks + row * 32 + cb)     = *(const v8h*)(src + CDIM);
            *(v8h*)(Ks + row * 32 + cb + 8) = *(const v8h*)(src + CDIM + 8);
#pragma unroll
            for (int c = 0; c < 16; c++)
                Vt[(cb + c) * 64 + row] = src[2 * CDIM + c];
        } else {
            v8h z = {};
            *(v8h*)(Qs + row * 32 + cb)     = z;
            *(v8h*)(Qs + row * 32 + cb + 8) = z;
            *(v8h*)(Ks + row * 32 + cb)     = z;
            *(v8h*)(Ks + row * 32 + cb + 8) = z;
#pragma unroll
            for (int c = 0; c < 16; c++)
                Vt[(cb + c) * 64 + row] = (_Float16)0.f;
        }
    }
    __syncthreads();

    // ---- S = Q (pre-scaled) @ K^T : wave handles M-tile `wave`, 4 N-tiles ----
    {
        v16h a = load_frag_a(Qs + wave * 16 * 32, 32, lane);
        int col0 = lane & 15;
        int r0   = wave * 16 + ((lane >> 4) << 3);
#pragma unroll
        for (int nt = 0; nt < 4; nt++) {
            v16h b = load_frag_b(Ks + nt * 16 * 32, 32, lane);
            v8f c = {};
            c = wmma_f16(a, b, c);
            int col = nt * 16 + col0;
#pragma unroll
            for (int v = 0; v < 8; v++) S[(r0 + v) * SROW + col] = c[v];
        }
    }
    __syncthreads();

    // ---- softmax over the 49 valid columns; rows >= 49 are zeros (finite) ----
    if (tid < 64) {
        float mx = -1e30f;
        for (int c = 0; c < NTOK; c++) mx = fmaxf(mx, S[tid * SROW + c]);
        float sum = 0.f;
        for (int c = 0; c < NTOK; c++) {
            float e = __expf(S[tid * SROW + c] - mx);
            S[tid * SROW + c] = e;
            sum += e;
        }
        float inv = 1.f / sum;
        for (int c = 0; c < NTOK; c++)
            P[tid * PROW + c] = (_Float16)(S[tid * SROW + c] * inv);
        for (int c = NTOK; c < 64; c++)
            P[tid * PROW + c] = (_Float16)0.f;
    }
    __syncthreads();

    // ---- Out = P @ V : wave handles M-tile `wave`, 2 N-tiles, K=64 ----
    {
        v8f oc[2] = {};
#pragma unroll
        for (int ks = 0; ks < 64; ks += 32) {
            v16h a = load_frag_a(P + wave * 16 * PROW + ks, PROW, lane);
#pragma unroll
            for (int nt = 0; nt < 2; nt++) {
                v16h b = load_frag_b(Vt + nt * 16 * 64 + ks, 64, lane);
                oc[nt] = wmma_f16(a, b, oc[nt]);
            }
        }
        int col0 = lane & 15;
        int r0   = wave * 16 + ((lane >> 4) << 3);
#pragma unroll
        for (int nt = 0; nt < 2; nt++)
#pragma unroll
            for (int v = 0; v < 8; v++) {
                int r = r0 + v;
                if (r < NTOK)
                    O[(rbase + r) * CDIM + h * HDIM + nt * 16 + col0] =
                        (_Float16)oc[nt][v];
            }
    }
}

// ---------------------------------------------------------------------------
// Kernel 4: proj GEMM (50176x512)@(512x512) + bias, window_reverse to fp32 out
// Epilogue: per 16-row sub-tile LDS repack (wave-private, barrier-free),
// bias folded at stage time, float4 coalesced stores, row remap once per row.
// ---------------------------------------------------------------------------
__global__ __launch_bounds__(128) void gemm_proj(
        const _Float16* __restrict__ A,   // attn out (M, 512)
        const _Float16* __restrict__ WT,  // (512, 512) N-major
        const float* __restrict__ bias,   // (512)
        float* __restrict__ out) {        // (16, 3136, 512)
    __shared__ __align__(16) float stage[4][16 * 64];    // 4KB per wave
    int lane = threadIdx.x & 31;
    int wave = threadIdx.x >> 5;
    int m0 = blockIdx.x * 128 + (wave >> 1) * 64;
    int n0 = blockIdx.y * 128 + (wave & 1) * 64;

    v8f acc[4][4] = {};
#pragma unroll 4
    for (int k = 0; k < CDIM; k += 32) {
        v16h a[4], b[4];
#pragma unroll
        for (int i = 0; i < 4; i++)
            a[i] = load_frag_a(A + (size_t)(m0 + i * 16) * CDIM + k, CDIM, lane);
#pragma unroll
        for (int j = 0; j < 4; j++)
            b[j] = load_frag_b(WT + (size_t)(n0 + j * 16) * CDIM + k, CDIM, lane);
#pragma unroll
        for (int i = 0; i < 4; i++)
#pragma unroll
            for (int j = 0; j < 4; j++)
                acc[i][j] = wmma_f16(a[i], b[j], acc[i][j]);
    }

    float* st  = stage[wave];
    int rowoff = (lane >> 4) << 3;
    int col    = lane & 15;
    float bj[4];
#pragma unroll
    for (int j = 0; j < 4; j++) bj[j] = bias[n0 + j * 16 + col];

#pragma unroll
    for (int i = 0; i < 4; i++) {
        // stage 16x64 f32 sub-tile with bias folded in
#pragma unroll
        for (int j = 0; j < 4; j++)
#pragma unroll
            for (int v = 0; v < 8; v++)
                st[(rowoff + v) * 64 + j * 16 + col] = acc[i][j][v] + bj[j];

        // coalesced float4 copy-out with window_reverse remap (per-row math)
#pragma unroll
        for (int it = 0; it < 8; it++) {
            int li = it * 32 + lane;      // 0..255
            int r  = li >> 4;             // row 0..15
            int ch = li & 15;             // float4 chunk
            float4 d = *(float4*)(st + r * 64 + ch * 4);
            int mm = m0 + i * 16 + r;     // window-token row
            int win = mm / NTOK;
            int t   = mm - win * NTOK;
            int b2  = win >> 6;
            int wrem = win & 63;
            int wh = wrem >> 3, ww = wrem & 7;
            int ii = t / 7, jj = t - ii * 7;
            int srow = (wh * 7 + ii) * 56 + ww * 7 + jj;
            *(float4*)(out + ((size_t)b2 * 3136 + srow) * CDIM + n0 + ch * 4) = d;
        }
    }
}

// ---------------------------------------------------------------------------
extern "C" void kernel_launch(void* const* d_in, const int* in_sizes, int n_in,
                              void* d_out, int out_size, void* d_ws, size_t ws_size,
                              hipStream_t stream) {
    const float* hs    = (const float*)d_in[0];   // (16, 3136, 512)
    const float* wqkv  = (const float*)d_in[1];   // (512, 1536)
    const float* wproj = (const float*)d_in[2];   // (512, 512)
    const float* bproj = (const float*)d_in[3];   // (512)
    float* out = (float*)d_out;

    // workspace carve-up (all sizes multiples of 256 bytes)
    char* ws = (char*)d_ws;
    _Float16* Xw    = (_Float16*)ws;                       // 50176*512  f16
    ws += (size_t)MROWS * CDIM * sizeof(_Float16);
    _Float16* QKVb  = (_Float16*)ws;                       // 50176*1536 f16
    ws += (size_t)MROWS * QKVDIM * sizeof(_Float16);
    _Float16* Attn  = (_Float16*)ws;                       // 50176*512  f16
    ws += (size_t)MROWS * CDIM * sizeof(_Float16);
    _Float16* wqkvT = (_Float16*)ws;                       // 1536*512   f16
    ws += (size_t)QKVDIM * CDIM * sizeof(_Float16);
    _Float16* wprojT = (_Float16*)ws;                      // 512*512    f16

    // 0) weight transpose/convert (+ folded q scale)
    prep_weights<<<(QKVDIM * CDIM + 255) / 256, 256, 0, stream>>>(
        wqkv, wproj, wqkvT, wprojT);

    // 1) window-partition gather to f16
    gather_windows<<<(MROWS * 64) / 256, 256, 0, stream>>>(hs, Xw);

    // 2) QKV GEMM: (50176 x 512) @ (512 x 1536)
    gemm_qkv<<<dim3(MROWS / 128, QKVDIM / 128), 128, 0, stream>>>(Xw, wqkvT, QKVb);

    // 3) attention per (window, head)
    attn_win<<<dim3(NWIN, NHEAD), 128, 0, stream>>>(QKVb, Attn);

    // 4) proj GEMM + bias + window_reverse
    gemm_proj<<<dim3(MROWS / 128, CDIM / 128), 128, 0, stream>>>(
        Attn, wprojT, bproj, out);
}
